// MultiHeadedAttention_20624432956239
// MI455X (gfx1250) — matrix operand order repaired and checked by reading, not compiled, not run on hardware
//
#include <hip/hip_runtime.h>
#include <hip/hip_bf16.h>

#define Bv 64
#define Tv 2048
#define Hv 512
#define NHv 8
#define Av 64

typedef __attribute__((ext_vector_type(16))) __bf16 v16bf;
typedef __attribute__((ext_vector_type(8)))  __bf16 v8bf;
typedef __attribute__((ext_vector_type(4)))  __bf16 v4bf;
typedef __attribute__((ext_vector_type(8)))  float  v8f;

__device__ __forceinline__ v16bf combine16(v8bf lo, v8bf hi) {
    return __builtin_shufflevector(lo, hi, 0,1,2,3,4,5,6,7,8,9,10,11,12,13,14,15);
}

// ---------------------------------------------------------------------------
// Pre-swizzle W1 [NH,H,A] fp32 -> bf16 in WMMA B-fragment layout:
//   W1s[((kb*32 + ntg)*32 + lane)*16 + j] = W1[head][kb*32 + (lane>>4)*16 + j][a]
//   ntg = global n-tile (0..31), col = ntg*16 + (lane&15), head=col>>6, a=col&63
// ---------------------------------------------------------------------------
__global__ __launch_bounds__(256) void swizzle_w1(const float* __restrict__ W1,
                                                  __bf16* __restrict__ W1s) {
    int idx = blockIdx.x * 256 + threadIdx.x;           // 0 .. 512*512-1
    int j    = idx & 15;
    int lane = (idx >> 4) & 31;
    int g    = idx >> 9;                                // kb*32 + ntg
    int ntg  = g & 31;
    int kb   = g >> 5;
    int k    = kb * 32 + (lane >> 4) * 16 + j;
    int col  = ntg * 16 + (lane & 15);
    int head = col >> 6;
    int a    = col & 63;
    W1s[idx] = (__bf16)W1[(head * Hv + k) * Av + a];
}

// W2 [NH,A,A] fp32 -> bf16, layout: [head][kb(2)][nt(4)][lane(32)][j(16)]
__global__ __launch_bounds__(256) void swizzle_w2(const float* __restrict__ W2,
                                                  __bf16* __restrict__ W2s) {
    int idx  = blockIdx.x * 256 + threadIdx.x;          // 0 .. 8*64*64-1
    int j    = idx & 15;
    int lane = (idx >> 4) & 31;
    int nt   = (idx >> 9) & 3;
    int kb   = (idx >> 11) & 1;
    int head = idx >> 12;
    int k    = kb * 32 + (lane >> 4) * 16 + j;
    int a    = nt * 16 + (lane & 15);
    W2s[idx] = (__bf16)W2[(head * Av + k) * Av + a];
}

// ---------------------------------------------------------------------------
// Scorer: per 32-row tile of [B*T] (two 16-row WMMA sub-tiles), compute
// logits[b, head, t] for all heads. 8 waves per block, wave == head.
// Each B-fragment load feeds 2 WMMAs (M-tiling halves VMEM per WMMA).
// ---------------------------------------------------------------------------
__global__ __launch_bounds__(256) void scorer_kernel(
    const float* __restrict__ x,
    const __bf16* __restrict__ W1s, const __bf16* __restrict__ W2s,
    const float* __restrict__ b1, const float* __restrict__ b2,
    const float* __restrict__ w3, const float* __restrict__ b3,
    float* __restrict__ logits)
{
    constexpr int XP = 520;  // padded x-tile row stride (halves): 1040B -> conflict-free
    constexpr int HP = 72;   // padded h1-tile row stride (halves): 144B -> conflict-free
    __shared__ __align__(16) __bf16 xt[32 * XP];          // 32 rows of x (bf16)
    __shared__ __align__(16) __bf16 ht[NHv][16 * HP];     // per-wave h1 tile (reused for both sub-tiles)

    const int tid  = threadIdx.x;
    const int head = tid >> 5;         // wave id == head
    const int lane = tid & 31;
    const int half = lane >> 4;
    const int l15  = lane & 15;

    const long rowbase = (long)blockIdx.x * 32;          // row in flattened [B*T]

    // cooperative load + fp32->bf16 convert of the 32x512 x tile (vectorized)
    {
        const float* xsrc = x + rowbase * Hv;
        for (int i = tid * 4; i < 32 * Hv; i += 256 * 4) {
            int r = i >> 9, c = i & 511;
            float4 v = *(const float4*)(xsrc + (long)r * Hv + c);
            v4bf pv = { (__bf16)v.x, (__bf16)v.y, (__bf16)v.z, (__bf16)v.w };
            *(v4bf*)&xt[r * XP + c] = pv;
        }
    }
    __syncthreads();

    // ---- GEMM1: h1[32x64] = x_tile[32x512] * W1[:, head*64 : head*64+64] ----
    v8f acc[2][4] = {};
    for (int kb = 0; kb < 16; ++kb) {
        const __bf16* ap0 = &xt[l15 * XP + kb * 32 + half * 16];
        const __bf16* ap1 = &xt[(16 + l15) * XP + kb * 32 + half * 16];
        v16bf a0 = combine16(*(const v8bf*)ap0, *(const v8bf*)(ap0 + 8));
        v16bf a1 = combine16(*(const v8bf*)ap1, *(const v8bf*)(ap1 + 8));
        #pragma unroll
        for (int t = 0; t < 4; ++t) {
            const __bf16* bp = W1s + ((size_t)((kb * 32 + head * 4 + t) * 32 + lane)) * 16;
            v16bf bfrag = combine16(*(const v8bf*)bp, *(const v8bf*)(bp + 8));
            acc[0][t] = __builtin_amdgcn_wmma_f32_16x16x32_bf16(
                false, a0, false, bfrag, (short)0, acc[0][t], false, false);
            acc[1][t] = __builtin_amdgcn_wmma_f32_16x16x32_bf16(
                false, a1, false, bfrag, (short)0, acc[1][t], false, false);
        }
    }

    // per-sub-tile epilogue + GEMM2 + logits (ht buffer reused; uniform barriers)
    float part[2][8];
    #pragma unroll
    for (int m = 0; m < 2; ++m) {
        // bias + relu, spill h1 sub-tile (bf16) to per-wave LDS region for re-layout
        #pragma unroll
        for (int t = 0; t < 4; ++t) {
            const int col = t * 16 + l15;
            const float bb = b1[head * Av + col];
            #pragma unroll
            for (int v = 0; v < 8; ++v) {
                float hv = fmaxf(acc[m][t][v] + bb, 0.0f);
                int row = v + half * 8;
                ht[head][row * HP + col] = (__bf16)hv;
            }
        }
        __syncthreads();

        // ---- GEMM2: h2[16x64] = h1[16x64] * W2[head] ----
        v8f c2[4] = {};
        #pragma unroll
        for (int kb = 0; kb < 2; ++kb) {
            const __bf16* ap = &ht[head][l15 * HP + kb * 32 + half * 16];
            v16bf afrag = combine16(*(const v8bf*)ap, *(const v8bf*)(ap + 8));
            #pragma unroll
            for (int t = 0; t < 4; ++t) {
                const __bf16* bp = W2s + ((size_t)(((head * 2 + kb) * 4 + t) * 32 + lane)) * 16;
                v16bf bfrag = combine16(*(const v8bf*)bp, *(const v8bf*)(bp + 8));
                c2[t] = __builtin_amdgcn_wmma_f32_16x16x32_bf16(
                    false, afrag, false, bfrag, (short)0, c2[t], false, false);
            }
        }
        __syncthreads();   // protect ht before next sub-tile overwrites it

        // ---- logits partials: relu(h2 + b2) . w3, reduced over 64 columns ----
        #pragma unroll
        for (int v = 0; v < 8; ++v) part[m][v] = 0.0f;
        #pragma unroll
        for (int t = 0; t < 4; ++t) {
            const int col = t * 16 + l15;
            const float bb = b2[head * Av + col];
            const float wv = w3[head * Av + col];
            #pragma unroll
            for (int v = 0; v < 8; ++v) {
                float h2 = fmaxf(c2[t][v] + bb, 0.0f);
                part[m][v] += h2 * wv;
            }
        }
        // reduce across the 16 lanes of each half (rows v+8*half)
        #pragma unroll
        for (int v = 0; v < 8; ++v) {
            float s = part[m][v];
            #pragma unroll
            for (int msk = 1; msk < 16; msk <<= 1) s += __shfl_xor(s, msk, 32);
            part[m][v] = s;
        }
    }

    if (l15 == 0) {
        const int b     = (int)(rowbase >> 11);          // T = 2048
        const int tbase = (int)(rowbase & 2047);
        const float bias3 = b3[head];
        #pragma unroll
        for (int m = 0; m < 2; ++m) {
            #pragma unroll
            for (int v = 0; v < 8; ++v) {
                int trow = tbase + m * 16 + v + half * 8;
                logits[((long)(b * NHv + head)) * Tv + trow] = part[m][v] + bias3;
            }
        }
    }
}

// ---------------------------------------------------------------------------
// Softmax over T, in place. One block per (b, head) row.
// ---------------------------------------------------------------------------
__global__ __launch_bounds__(256) void softmax_kernel(float* __restrict__ logits) {
    __shared__ float red[256];
    const int tid = threadIdx.x;
    float* p = logits + (long)blockIdx.x * Tv;

    float vals[8];
    float mx = -INFINITY;
    #pragma unroll
    for (int i = 0; i < 8; ++i) { vals[i] = p[tid + i * 256]; mx = fmaxf(mx, vals[i]); }
    red[tid] = mx; __syncthreads();
    for (int s = 128; s > 0; s >>= 1) {
        if (tid < s) red[tid] = fmaxf(red[tid], red[tid + s]);
        __syncthreads();
    }
    mx = red[0]; __syncthreads();

    float sum = 0.0f;
    #pragma unroll
    for (int i = 0; i < 8; ++i) { vals[i] = __expf(vals[i] - mx); sum += vals[i]; }
    red[tid] = sum; __syncthreads();
    for (int s = 128; s > 0; s >>= 1) {
        if (tid < s) red[tid] += red[tid + s];
        __syncthreads();
    }
    const float inv = 1.0f / red[0];
    #pragma unroll
    for (int i = 0; i < 8; ++i) p[tid + i * 256] = vals[i] * inv;
}

// attention_weights[b,t] = mean over heads / sqrt(H)
__global__ __launch_bounds__(256) void mean_kernel(const float* __restrict__ w,
                                                   float* __restrict__ attn) {
    long i = (long)blockIdx.x * 256 + threadIdx.x;      // over B*T
    int b = (int)(i >> 11), t = (int)(i & 2047);
    float s = 0.0f;
    #pragma unroll
    for (int n = 0; n < NHv; ++n) s += w[((long)(b * NHv + n)) * Tv + t];
    attn[i] = s * 0.005524271728019903f;                // 1 / (8 * sqrt(512))
}

// ---------------------------------------------------------------------------
// Pooling: partial[tc][b][n][h] = sum over 256 t of w[b,n,t] * x[b,t,h]
// Deterministic (no atomics). Each x element read exactly once across the grid.
// ---------------------------------------------------------------------------
__global__ __launch_bounds__(512) void pool_kernel(const float* __restrict__ x,
                                                   const float* __restrict__ w,
                                                   float* __restrict__ part) {
    __shared__ float wt[NHv][256];
    const int tc = blockIdx.x, b = blockIdx.y;
    const int tid = threadIdx.x;
    for (int i = tid; i < NHv * 256; i += 512) {
        int n = i >> 8, tt = i & 255;
        wt[n][tt] = w[((long)(b * NHv + n)) * Tv + tc * 256 + tt];
    }
    __syncthreads();

    const int h = tid;                                   // 512 threads == H columns
    const float* xp = x + ((long)b * Tv + tc * 256) * Hv + h;
    float acc[NHv] = {0,0,0,0,0,0,0,0};
    for (int t = 0; t < 256; ++t) {
        float xv = xp[(long)t * Hv];
        #pragma unroll
        for (int n = 0; n < NHv; ++n) acc[n] += wt[n][t] * xv;
    }
    float* pp = part + (((long)tc * Bv + b) * NHv) * Hv + h;
    #pragma unroll
    for (int n = 0; n < NHv; ++n) pp[(long)n * Hv] = acc[n];
}

__global__ __launch_bounds__(256) void ctx_reduce_kernel(const float* __restrict__ part,
                                                         float* __restrict__ out) {
    long i = (long)blockIdx.x * 256 + threadIdx.x;      // over B*NH*H = 262144
    float s = 0.0f;
    #pragma unroll
    for (int tc = 0; tc < 8; ++tc) s += part[(long)tc * (Bv * NHv * Hv) + i];
    out[i] = s;
}

// ---------------------------------------------------------------------------
extern "C" void kernel_launch(void* const* d_in, const int* in_sizes, int n_in,
                              void* d_out, int out_size, void* d_ws, size_t ws_size,
                              hipStream_t stream) {
    const float* x  = (const float*)d_in[0];
    const float* W1 = (const float*)d_in[1];
    const float* b1 = (const float*)d_in[2];
    const float* W2 = (const float*)d_in[3];
    const float* b2 = (const float*)d_in[4];
    const float* w3 = (const float*)d_in[5];
    const float* b3 = (const float*)d_in[6];

    float* ctx_out  = (float*)d_out;                             // [B, NH*H]
    float* attn_out = (float*)d_out + (long)Bv * NHv * Hv;       // [B, T]

    char* ws = (char*)d_ws;
    __bf16* W1s   = (__bf16*)ws;                                   // 512 KB
    __bf16* W2s   = (__bf16*)(ws + (512 << 10));                   //  64 KB
    float*  logit = (float*)(ws + (512 << 10) + (64 << 10));       //   4 MB (logits -> weights in place)
    float*  part  = (float*)(ws + (512 << 10) + (64 << 10) + (4 << 20)); // 8 MB

    swizzle_w1<<<(Hv * NHv * Av) / 256, 256, 0, stream>>>(W1, W1s);
    swizzle_w2<<<(NHv * Av * Av) / 256, 256, 0, stream>>>(W2, W2s);

    scorer_kernel<<<(Bv * Tv) / 32, 256, 0, stream>>>(x, W1s, W2s, b1, b2, w3, b3, logit);

    softmax_kernel<<<Bv * NHv, 256, 0, stream>>>(logit);
    mean_kernel<<<(Bv * Tv) / 256, 256, 0, stream>>>(logit, attn_out);

    pool_kernel<<<dim3(8, Bv), 512, 0, stream>>>(x, logit, part);
    ctx_reduce_kernel<<<(Bv * NHv * Hv) / 256, 256, 0, stream>>>(part, ctx_out);
}
